// SelfAttention_25125558681714
// MI455X (gfx1250) — matrix operand order repaired and checked by reading, not compiled, not run on hardware
//
#include <hip/hip_runtime.h>
#include <hip/hip_bf16.h>

#define B_  8
#define C_  512
#define L_  4096
#define DK_ 64
#define DV_ 256
#define LP_ 2048   // L/2 after maxpool

typedef __bf16 bf16;
typedef __attribute__((ext_vector_type(16))) __bf16 v16bf;
typedef __attribute__((ext_vector_type(8)))  float  v8f;

__device__ inline v8f wmma_bf16(v16bf a, v16bf b, v8f c) {
  // D = A(16x32 bf16) * B(32x16 bf16) + C(16x16 f32)
  return __builtin_amdgcn_wmma_f32_16x16x32_bf16(false, a, false, b, (short)0, c,
                                                 false, false);
}

// ---- fragment builders (ISA 7.12.2 layouts, wave32) -----------------------
// A: 16x32 bf16. lane&15 = row M; lanes>=16 hold K+8 in each VGPR-pair slot.
// VGPR j<4 pair p -> K = 2j+p (+koff); j>=4 -> K = 16+2(j-4)+p (+koff).
template <typename T>
__device__ inline v16bf load_a_frag(const T* __restrict__ A, int lda, int lane) {
  v16bf f;
  const int row  = lane & 15;
  const int koff = (lane >> 4) * 8;
  const T* rp = A + (long)row * lda + koff;
#pragma unroll
  for (int j = 0; j < 8; ++j) {
    const int kbase = (j < 4) ? (2 * j) : (16 + 2 * (j - 4));
#pragma unroll
    for (int p = 0; p < 2; ++p)
      f[2 * j + p] = (bf16)rp[kbase + p];
  }
  return f;
}

// B: 32x16 bf16. lane&15 = col N; elem i<8 -> K=8h+i, i>=8 -> K=16+8h+(i-8) (h=lane>>4).
// Source row-major [K][N] with leading dim ldb (N contiguous -> coalesced lanes).
template <typename T>
__device__ inline v16bf load_b_frag(const T* __restrict__ Bsrc, long ldb, int lane) {
  v16bf f;
  const int col  = lane & 15;
  const int koff = (lane >> 4) * 8;
  const T* cp = Bsrc + (long)koff * ldb + col;
#pragma unroll
  for (int i = 0; i < 16; ++i)
    f[i] = (bf16)cp[(long)(i < 8 ? i : i + 8) * ldb];
  return f;
}

// ---- K1a: q = W_q @ x, store bf16 [B][DK][L] ------------------------------
__global__ __launch_bounds__(256) void proj_q_kernel(const float* __restrict__ wq,
                                                     const float* __restrict__ x,
                                                     bf16* __restrict__ qb) {
  const int b  = blockIdx.y;
  const int l0 = blockIdx.x * 64;
  const int tid = threadIdx.x, wave = tid >> 5, lane = tid & 31;
  const float* X = x + (long)b * C_ * L_;
  const int t0  = wave * 2;
  const int mi  = t0 >> 2;            // 16-row tile of DK (0..3)
  const int ni0 = t0 & 3, ni1 = ni0 + 1;
  v8f acc0 = {}, acc1 = {};
  for (int k0 = 0; k0 < C_; k0 += 32) {
    v16bf a  = load_a_frag(wq + (long)(mi * 16) * C_ + k0, C_, lane);
    v16bf b0 = load_b_frag(X + (long)k0 * L_ + l0 + ni0 * 16, L_, lane);
    v16bf b1 = load_b_frag(X + (long)k0 * L_ + l0 + ni1 * 16, L_, lane);
    acc0 = wmma_bf16(a, b0, acc0);
    acc1 = wmma_bf16(a, b1, acc1);
  }
  const int colb = lane & 15, hi = (lane >> 4) * 8;
#pragma unroll
  for (int r = 0; r < 8; ++r) {
    const int d = mi * 16 + r + hi;
    qb[((long)b * DK_ + d) * L_ + l0 + ni0 * 16 + colb] = (bf16)acc0[r];
    qb[((long)b * DK_ + d) * L_ + l0 + ni1 * 16 + colb] = (bf16)acc1[r];
  }
}

// ---- K1b: k = W_k @ x, maxpool2 + transpose -> kpT bf16 [B][LP][DK] -------
__global__ __launch_bounds__(256) void proj_k_kernel(const float* __restrict__ wk,
                                                     const float* __restrict__ x,
                                                     bf16* __restrict__ kpT) {
  __shared__ float tile[64][65];      // [d][l], padded
  const int b  = blockIdx.y;
  const int l0 = blockIdx.x * 64;
  const int tid = threadIdx.x, wave = tid >> 5, lane = tid & 31;
  const float* X = x + (long)b * C_ * L_;
  const int t0  = wave * 2;
  const int mi  = t0 >> 2;
  const int ni0 = t0 & 3, ni1 = ni0 + 1;
  v8f acc0 = {}, acc1 = {};
  for (int k0 = 0; k0 < C_; k0 += 32) {
    v16bf a  = load_a_frag(wk + (long)(mi * 16) * C_ + k0, C_, lane);
    v16bf b0 = load_b_frag(X + (long)k0 * L_ + l0 + ni0 * 16, L_, lane);
    v16bf b1 = load_b_frag(X + (long)k0 * L_ + l0 + ni1 * 16, L_, lane);
    acc0 = wmma_bf16(a, b0, acc0);
    acc1 = wmma_bf16(a, b1, acc1);
  }
  const int colb = lane & 15, hi = (lane >> 4) * 8;
#pragma unroll
  for (int r = 0; r < 8; ++r) {
    tile[mi * 16 + r + hi][ni0 * 16 + colb] = acc0[r];
    tile[mi * 16 + r + hi][ni1 * 16 + colb] = acc1[r];
  }
  __syncthreads();
  const int m0 = l0 >> 1;
  for (int idx = tid; idx < 32 * 64; idx += 256) {
    const int m = idx >> 6, d = idx & 63;
    const float v = fmaxf(tile[d][2 * m], tile[d][2 * m + 1]);
    kpT[((long)b * LP_ + m0 + m) * DK_ + d] = (bf16)v;
  }
}

// ---- K1c: v = W_v @ x, maxpool2 -> vp bf16 [B][DV][LP] --------------------
__global__ __launch_bounds__(256) void proj_v_kernel(const float* __restrict__ wv,
                                                     const float* __restrict__ x,
                                                     bf16* __restrict__ vp) {
  __shared__ float tile[64][65];      // [row][l]
  const int b  = blockIdx.z;
  const int r0 = blockIdx.y * 64;
  const int l0 = blockIdx.x * 64;
  const int tid = threadIdx.x, wave = tid >> 5, lane = tid & 31;
  const float* X = x + (long)b * C_ * L_;
  const int t0  = wave * 2;
  const int mi  = t0 >> 2;
  const int ni0 = t0 & 3, ni1 = ni0 + 1;
  v8f acc0 = {}, acc1 = {};
  for (int k0 = 0; k0 < C_; k0 += 32) {
    v16bf a  = load_a_frag(wv + (long)(r0 + mi * 16) * C_ + k0, C_, lane);
    v16bf b0 = load_b_frag(X + (long)k0 * L_ + l0 + ni0 * 16, L_, lane);
    v16bf b1 = load_b_frag(X + (long)k0 * L_ + l0 + ni1 * 16, L_, lane);
    acc0 = wmma_bf16(a, b0, acc0);
    acc1 = wmma_bf16(a, b1, acc1);
  }
  const int colb = lane & 15, hi = (lane >> 4) * 8;
#pragma unroll
  for (int r = 0; r < 8; ++r) {
    tile[mi * 16 + r + hi][ni0 * 16 + colb] = acc0[r];
    tile[mi * 16 + r + hi][ni1 * 16 + colb] = acc1[r];
  }
  __syncthreads();
  const int m0 = l0 >> 1;
  for (int idx = tid; idx < 64 * 32; idx += 256) {
    const int rr = idx >> 5, m = idx & 31;
    const float v = fmaxf(tile[rr][2 * m], tile[rr][2 * m + 1]);
    vp[((long)b * DV_ + r0 + rr) * LP_ + m0 + m] = (bf16)v;
  }
}

// ---- K2: raw scores[b][m][l] = sum_d kpT[m][d] * q[d][l] ------------------
__global__ __launch_bounds__(256) void scores_kernel(const bf16* __restrict__ kpT,
                                                     const bf16* __restrict__ qb,
                                                     float* __restrict__ attn) {
  const int b  = blockIdx.z;
  const int m0 = blockIdx.y * 128;
  const int l0 = blockIdx.x * 128;
  const int tid = threadIdx.x, wave = tid >> 5, lane = tid & 31;
  const bf16* Kp = kpT + (long)b * LP_ * DK_;
  const bf16* Q  = qb  + (long)b * DK_ * L_;
  float* S = attn + (long)b * LP_ * L_;
  v8f acc[8] = {};
  const int mi = wave;                // each wave owns one 16-row stripe
  for (int k0 = 0; k0 < DK_; k0 += 32) {
    v16bf a = load_a_frag(Kp + (long)(m0 + mi * 16) * DK_ + k0, DK_, lane);
#pragma unroll
    for (int ni = 0; ni < 8; ++ni) {
      v16bf bf = load_b_frag(Q + (long)k0 * L_ + l0 + ni * 16, L_, lane);
      acc[ni] = wmma_bf16(a, bf, acc[ni]);
    }
  }
  const int colb = lane & 15, hi = (lane >> 4) * 8;
#pragma unroll
  for (int ni = 0; ni < 8; ++ni) {
    const int col = l0 + ni * 16 + colb;
#pragma unroll
    for (int r = 0; r < 8; ++r)
      S[(long)(m0 + mi * 16 + r + hi) * L_ + col] = acc[ni][r];
  }
}

// ---- K3: softmax over m (axis of length LP) for 32 columns per WG ---------
// Whole [2048 x 32] slab lives in 256KB dynamic LDS: one read, one write.
__global__ __launch_bounds__(256) void softmax_kernel(float* __restrict__ attn) {
  extern __shared__ float sm[];       // LP_*32 floats = 256KB
  __shared__ float red[8][33];
  __shared__ float colv[32];
  const int b  = blockIdx.y;
  const int l0 = blockIdx.x * 32;
  float* P = attn + (long)b * LP_ * L_ + l0;
  const int tid = threadIdx.x;
  for (int idx = tid; idx < LP_ * 32; idx += 256) {
    const int m = idx >> 5, c = idx & 31;
    sm[idx] = P[(long)m * L_ + c];
  }
  __syncthreads();
  const int c = tid & 31, g8 = tid >> 5;
  float mx = -3.0e38f;
  for (int i = 0; i < LP_ / 8; ++i)
    mx = fmaxf(mx, sm[(g8 * (LP_ / 8) + i) * 32 + c]);
  red[g8][c] = mx;
  __syncthreads();
  if (tid < 32) {
    float m2 = red[0][tid];
#pragma unroll
    for (int j = 1; j < 8; ++j) m2 = fmaxf(m2, red[j][tid]);
    colv[tid] = m2;
  }
  __syncthreads();
  const float cmax = colv[c];
  float s = 0.f;
  for (int i = 0; i < LP_ / 8; ++i) {
    const int off = (g8 * (LP_ / 8) + i) * 32 + c;
    const float e = __expf(sm[off] - cmax);
    sm[off] = e;
    s += e;
  }
  red[g8][c] = s;
  __syncthreads();
  if (tid < 32) {
    float t = 0.f;
#pragma unroll
    for (int j = 0; j < 8; ++j) t += red[j][tid];
    colv[tid] = 1.0f / t;
  }
  __syncthreads();
  for (int idx = tid; idx < LP_ * 32; idx += 256) {
    const int m = idx >> 5, cc = idx & 31;
    P[(long)m * L_ + cc] = sm[idx] * colv[cc];
  }
}

// ---- K4: out1[b][r][l] = sum_m vp[r][m] * attn[m][l], bf16 out ------------
// One WG owns ALL 256 V-rows for a 64-col stripe: attn streamed exactly once.
// 16 M-tiles x 4 N-tiles = 64 WMMA tiles; wave owns 2 M-tiles x 4 N-tiles.
__global__ __launch_bounds__(256) void av_kernel(const bf16* __restrict__ vp,
                                                 const float* __restrict__ attn,
                                                 bf16* __restrict__ out1) {
  const int b  = blockIdx.z;
  const int l0 = blockIdx.x * 64;
  const int tid = threadIdx.x, wave = tid >> 5, lane = tid & 31;
  const bf16*  V = vp + (long)b * DV_ * LP_;
  const float* P = attn + (long)b * LP_ * L_;
  v8f acc[2][4] = {};
  const int mi0 = wave * 2;           // wave owns M-tiles mi0, mi0+1 (of 16)
  for (int k0 = 0; k0 < LP_; k0 += 32) {
    if (k0 + 32 < LP_)
      __builtin_prefetch(P + (long)(k0 + 32) * L_ + l0 + lane * 4, 0, 1);
    v16bf a0 = load_a_frag(V + (long)(mi0 * 16) * LP_ + k0, LP_, lane);
    v16bf a1 = load_a_frag(V + (long)(mi0 * 16 + 16) * LP_ + k0, LP_, lane);
#pragma unroll
    for (int q = 0; q < 4; ++q) {
      v16bf bf = load_b_frag(P + (long)k0 * L_ + l0 + q * 16, L_, lane);
      acc[0][q] = wmma_bf16(a0, bf, acc[0][q]);
      acc[1][q] = wmma_bf16(a1, bf, acc[1][q]);
    }
  }
  const int colb = lane & 15, hi = (lane >> 4) * 8;
#pragma unroll
  for (int j = 0; j < 2; ++j)
#pragma unroll
    for (int q = 0; q < 4; ++q) {
      const int col = l0 + q * 16 + colb;
#pragma unroll
      for (int r = 0; r < 8; ++r)
        out1[((long)b * DV_ + (mi0 + j) * 16 + r + hi) * L_ + col] =
            (bf16)acc[j][q][r];
    }
}

// ---- K5: y = gamma * (W_f @ out1) + x -------------------------------------
// 256-row stripes (grid y = 2): out1 streamed only twice.
__global__ __launch_bounds__(256) void final_kernel(const float* __restrict__ wf,
                                                    const bf16* __restrict__ out1,
                                                    const float* __restrict__ x,
                                                    const float* __restrict__ gamma,
                                                    float* __restrict__ y) {
  const int b  = blockIdx.z;
  const int f0 = blockIdx.y * 256;
  const int l0 = blockIdx.x * 64;
  const int tid = threadIdx.x, wave = tid >> 5, lane = tid & 31;
  const bf16* O = out1 + (long)b * DV_ * L_;
  v8f acc[2][4] = {};
  const int mi0 = wave * 2;           // wave owns M-tiles mi0, mi0+1 (of 16)
  for (int k0 = 0; k0 < DV_; k0 += 32) {
    v16bf a0 = load_a_frag(wf + (long)(f0 + mi0 * 16) * DV_ + k0, DV_, lane);
    v16bf a1 = load_a_frag(wf + (long)(f0 + mi0 * 16 + 16) * DV_ + k0, DV_, lane);
#pragma unroll
    for (int q = 0; q < 4; ++q) {
      v16bf bf = load_b_frag(O + (long)k0 * L_ + l0 + q * 16, L_, lane);
      acc[0][q] = wmma_bf16(a0, bf, acc[0][q]);
      acc[1][q] = wmma_bf16(a1, bf, acc[1][q]);
    }
  }
  const float g = gamma[0];
  const int colb = lane & 15, hi = (lane >> 4) * 8;
#pragma unroll
  for (int j = 0; j < 2; ++j)
#pragma unroll
    for (int q = 0; q < 4; ++q) {
      const int col = l0 + q * 16 + colb;
#pragma unroll
      for (int r = 0; r < 8; ++r) {
        const long off =
            ((long)b * C_ + f0 + (mi0 + j) * 16 + r + hi) * L_ + col;
        y[off] = g * acc[j][q][r] + x[off];
      }
    }
}

// ---------------------------------------------------------------------------
extern "C" void kernel_launch(void* const* d_in, const int* in_sizes, int n_in,
                              void* d_out, int out_size, void* d_ws, size_t ws_size,
                              hipStream_t stream) {
  const float* x     = (const float*)d_in[0];
  const float* w_q   = (const float*)d_in[1];
  const float* w_k   = (const float*)d_in[2];
  const float* w_v   = (const float*)d_in[3];
  const float* w_f   = (const float*)d_in[4];
  const float* gamma = (const float*)d_in[5];

  float* y    = (float*)d_out;                       // [B][C][L]
  float* attn = y + (size_t)B_ * C_ * L_;            // [B][LP][L]

  char* ws = (char*)d_ws;                            // ~30 MB used
  bf16* qb   = (bf16*)(ws);                          //  4 MB  [B][DK][L]
  bf16* kpT  = (bf16*)(ws + (4u  << 20));            //  2 MB  [B][LP][DK]
  bf16* vp   = (bf16*)(ws + (6u  << 20));            //  8 MB  [B][DV][LP]
  bf16* out1 = (bf16*)(ws + (14u << 20));            // 16 MB  [B][DV][L]

  proj_q_kernel<<<dim3(L_ / 64, B_), 256, 0, stream>>>(w_q, x, qb);
  proj_k_kernel<<<dim3(L_ / 64, B_), 256, 0, stream>>>(w_k, x, kpT);
  proj_v_kernel<<<dim3(L_ / 64, DV_ / 64, B_), 256, 0, stream>>>(w_v, x, vp);
  scores_kernel<<<dim3(L_ / 128, LP_ / 128, B_), 256, 0, stream>>>(kpT, qb, attn);
  softmax_kernel<<<dim3(L_ / 32, B_), 256, LP_ * 32 * sizeof(float), stream>>>(attn);
  av_kernel<<<dim3(L_ / 64, 1, B_), 256, 0, stream>>>(vp, attn, out1);
  final_kernel<<<dim3(L_ / 64, C_ / 256, B_), 256, 0, stream>>>(w_f, out1, x, gamma, y);
}
